// digitrec_sw_77635828842790
// MI455X (gfx1250) — compile-verified, matrix operand order checked by reading, not run on hardware
//
#include <hip/hip_runtime.h>
#include <stdint.h>

typedef __attribute__((ext_vector_type(8))) int v8i;

#define NUM_TRAIN  50000
#define NUM_TEST   4096
#define DW         256          // binary elements per row
#define ROW_BYTES  256          // packed bytes per row (int8)
#define CLASS_SIZE 5000
#define N_TILES    (NUM_TRAIN / 16)   // 3125 (exact)
#define WAVES      8

__device__ __forceinline__ unsigned umin32(unsigned a, unsigned b) { return a < b ? a : b; }

// v_med3_u32: median of three (sorted-insert workhorse, 1 VOP3 vs min+max pair)
__device__ __forceinline__ unsigned umed3(unsigned a, unsigned b, unsigned c) {
  unsigned d;
  asm("v_med3_u32 %0, %1, %2, %3" : "=v"(d) : "v"(a), "v"(b), "v"(c));
  return d;
}

// v_mad_i32_i24: d = a*b + c  (|a| < 2^23). One VALU for kb - (inner << 17).
__device__ __forceinline__ unsigned mad_i24(int a, int b_sgpr, unsigned c) {
  unsigned d;
  asm("v_mad_i32_i24 %0, %1, %2, %3" : "=v"(d) : "v"(a), "s"(b_sgpr), "v"(c));
  return d;
}

// ---------------------------------------------------------------------------
// Phase 1: pack int32 {0,1} rows into WMMA-operand-ready int8 layout.
// Output dword order per row: for ko in 0..3, for half in 0..1, the 8 dwords a
// lane needs for V_WMMA_*_IU8 operand regs v0..v7 (src dword offsets
// {0,1,4,5,8,9,12,13} within the K=64 block) — so the main loop can fetch each
// operand K-step as two contiguous global_load_b128s.
// Also emits keyBase[j] = ((row_sum + 512) << 16) | j  (packed sort key base).
// ---------------------------------------------------------------------------
__global__ void pack_rows_kernel(const int* __restrict__ src,
                                 uint8_t* __restrict__ dst,
                                 unsigned* __restrict__ keyBase,
                                 int nrows) {
  int j = blockIdx.x * blockDim.x + threadIdx.x;
  if (j >= nrows) return;
  const int* r = src + (size_t)j * DW;
  unsigned* o32 = (unsigned*)(dst + (size_t)j * ROW_BYTES);
  unsigned sum = 0;
  int d = 0;
#pragma unroll
  for (int ko = 0; ko < 4; ++ko)
#pragma unroll
    for (int half = 0; half < 2; ++half)
#pragma unroll
      for (int q = 0; q < 8; ++q) {
        int g = ko * 16 + half * 2 + ((q >> 1) * 4 + (q & 1));  // source dword
        unsigned v0 = (unsigned)r[4 * g + 0];
        unsigned v1 = (unsigned)r[4 * g + 1];
        unsigned v2 = (unsigned)r[4 * g + 2];
        unsigned v3 = (unsigned)r[4 * g + 3];
        sum += v0 + v1 + v2 + v3;
        o32[d++] = v0 | (v1 << 8) | (v2 << 16) | (v3 << 24);
      }
  if (keyBase) keyBase[j] = ((sum + 512u) << 16) | (unsigned)j;
}

// --- helpers ---------------------------------------------------------------
__device__ __forceinline__ v8i load_op32(const uint8_t* p) {
  const int4* q = (const int4*)p;
  int4 lo = q[0];
  int4 hi = q[1];
  v8i v;
  v[0] = lo.x; v[1] = lo.y; v[2] = lo.z; v[3] = lo.w;
  v[4] = hi.x; v[5] = hi.y; v[6] = hi.z; v[7] = hi.w;
  return v;
}

// 16x16 tile, K=256 via 4 chained V_WMMA_I32_16X16X64_IU8
__device__ __forceinline__ v8i mm_k256(const v8i areg[4], const uint8_t* bp) {
  v8i c = {};
#pragma unroll
  for (int ko = 0; ko < 4; ++ko) {
    v8i b = load_op32(bp + ko * 64);
    c = __builtin_amdgcn_wmma_i32_16x16x64_iu8(false, areg[ko], false, b, c,
                                               false, false);
  }
  return c;
}

// Sorted-insert x into ascending (t0,t1,t2), dropping the max: 3 VALU.
__device__ __forceinline__ void ins3(unsigned x, unsigned& t0, unsigned& t1,
                                     unsigned& t2) {
  unsigned n0 = umin32(t0, x);
  unsigned n1 = umed3(x, t0, t1);
  unsigned n2 = umed3(x, t1, t2);
  t0 = n0; t1 = n1; t2 = n2;
}

// Insert 8 packed keys (one per C row) into per-row top-3 lists: 4 VALU each.
__device__ __forceinline__ void insert8(unsigned kb, int negsh, const v8i& c,
                                        unsigned t0[8], unsigned t1[8],
                                        unsigned t2[8]) {
#pragma unroll
  for (int rr = 0; rr < 8; ++rr) {
    // packed = ((tr_sum - 2*inner + 512) << 16) | j == inner*(-2^17) + kb
    unsigned x = mad_i24(c[rr], negsh, kb);
    ins3(x, t0[rr], t1[rr], t2[rr]);
  }
}

// ---------------------------------------------------------------------------
// Main fused kernel: per 16-row test tile (one block), sweep all 3125 train
// tiles with IU8 WMMA. Software-pipelined: each iteration inserts the
// *previous* pair's accumulators while the current pair's WMMA chains run, so
// in steady state no VALU reads a same-iteration WMMA destination (no hazard
// NOPs) and the XDL pipe overlaps the top-3 bookkeeping.
// ---------------------------------------------------------------------------
__global__ void __launch_bounds__(256)
knn_wmma_kernel(const uint8_t* __restrict__ testB,
                const uint8_t* __restrict__ trainB,
                const unsigned* __restrict__ keyBase,
                float* __restrict__ out) {
  __shared__ unsigned cand[16 * WAVES * 16 * 3];  // 24 KB
  __shared__ unsigned cand2[16 * 8 * 3];          // 1.5 KB (stage-2 partials)

  const int tid   = threadIdx.x;
  const int wave  = tid >> 5;
  const int lane  = tid & 31;
  const int half  = lane >> 4;   // C rows 0-7 (half=0) vs 8-15 (half=1)
  const int idx16 = lane & 15;   // C column within tile / operand row-col id
  const int negsh = -131072;     // -(1<<17), kept in an SGPR for v_mad_i32_i24

  // --- Load this wave's A operand (16 test rows × K=256) once: 32 VGPRs ----
  const int testRow = blockIdx.x * 16 + idx16;
  const uint8_t* ap = testB + (size_t)testRow * ROW_BYTES + half * 32;
  v8i areg[4];
#pragma unroll
  for (int ko = 0; ko < 4; ++ko) areg[ko] = load_op32(ap + ko * 64);

  // Per-lane sorted top-3 (ascending) of packed keys, for 8 C rows each.
  unsigned t0[8], t1[8], t2[8];
#pragma unroll
  for (int rr = 0; rr < 8; ++rr) { t0[rr] = ~0u; t1[rr] = ~0u; t2[rr] = ~0u; }

  const uint8_t* trB = trainB + half * 32;

  // --- Prologue: compute first pair (tiles wave, wave+WAVES) ---------------
  int n = wave;
  unsigned kb0 = keyBase[n * 16 + idx16];
  unsigned kb1 = keyBase[(n + WAVES) * 16 + idx16];
  v8i p0 = mm_k256(areg, trB + (size_t)(n * 16 + idx16) * ROW_BYTES);
  v8i p1 = mm_k256(areg, trB + (size_t)((n + WAVES) * 16 + idx16) * ROW_BYTES);
  n += 2 * WAVES;

  // --- Steady state: insert previous pair, compute next pair --------------
#pragma unroll 1
  for (; n + WAVES < N_TILES; n += 2 * WAVES) {
    insert8(kb0, negsh, p0, t0, t1, t2);   // reads prior-iteration WMMA D
    insert8(kb1, negsh, p1, t0, t1, t2);
    const int j0 = n * 16 + idx16;
    const int j1 = (n + WAVES) * 16 + idx16;
    kb0 = keyBase[j0];                      // redefined after last use
    kb1 = keyBase[j1];
    p0 = mm_k256(areg, trB + (size_t)j0 * ROW_BYTES);
    p1 = mm_k256(areg, trB + (size_t)j1 * ROW_BYTES);
  }

  // --- Epilogue: drain pipeline, then wave-uniform remainder tile ---------
  insert8(kb0, negsh, p0, t0, t1, t2);
  insert8(kb1, negsh, p1, t0, t1, t2);
  if (n < N_TILES) {                        // EXEC stays full (wave-uniform)
    const int j0 = n * 16 + idx16;
    const unsigned kr = keyBase[j0];
    v8i cr = mm_k256(areg, trB + (size_t)j0 * ROW_BYTES);
    insert8(kr, negsh, cr, t0, t1, t2);
  }

  // --- Dump per-(wave,lane) top-3 to LDS ----------------------------------
#pragma unroll
  for (int rr = 0; rr < 8; ++rr) {
    const int row = half * 8 + rr;
    const int base = ((row * WAVES + wave) * 16 + idx16) * 3;
    cand[base + 0] = t0[rr];
    cand[base + 1] = t1[rr];
    cand[base + 2] = t2[rr];
  }
  __syncthreads();

  // --- Stage 1: 8 threads per row, 48 candidates each ---------------------
  if (tid < 128) {
    const int row = tid >> 3;
    const int seg = tid & 7;
    unsigned m0 = ~0u, m1 = ~0u, m2 = ~0u;
    const unsigned* c0 = &cand[row * (WAVES * 16 * 3) + seg * 48];
#pragma unroll 4
    for (int s = 0; s < 48; ++s) ins3(c0[s], m0, m1, m2);
    const int b2 = (row * 8 + seg) * 3;
    cand2[b2 + 0] = m0;
    cand2[b2 + 1] = m1;
    cand2[b2 + 2] = m2;
  }
  __syncthreads();

  // --- Stage 2: one thread per row, 24 candidates; vote; argmax -----------
  if (tid < 16) {
    const int row = tid;
    unsigned m0 = ~0u, m1 = ~0u, m2 = ~0u;
    const unsigned* c0 = &cand2[row * 24];
#pragma unroll
    for (int s = 0; s < 24; ++s) ins3(c0[s], m0, m1, m2);
    const int l0 = (int)(m0 & 0xFFFFu) / CLASS_SIZE;
    const int l1 = (int)(m1 & 0xFFFFu) / CLASS_SIZE;
    const int l2 = (int)(m2 & 0xFFFFu) / CLASS_SIZE;
    // Majority vote of 3; argmax-first-max => all-distinct ties pick min label.
    int win;
    if (l0 == l1 || l0 == l2)       win = l0;
    else if (l1 == l2)              win = l1;
    else                            win = l0 < l1 ? (l0 < l2 ? l0 : l2)
                                                  : (l1 < l2 ? l1 : l2);
    out[blockIdx.x * 16 + row] = (float)win;
  }
}

// ---------------------------------------------------------------------------
extern "C" void kernel_launch(void* const* d_in, const int* in_sizes, int n_in,
                              void* d_out, int out_size, void* d_ws, size_t ws_size,
                              hipStream_t stream) {
  const int* training_set = (const int*)d_in[0];  // [50000, 256] int32 0/1
  const int* test_set     = (const int*)d_in[1];  // [4096, 256]  int32 0/1
  float* out = (float*)d_out;                     // [4096]

  uint8_t* ws = (uint8_t*)d_ws;
  uint8_t* trainB = ws;                                                  // 12.8 MB
  uint8_t* testB  = ws + (size_t)NUM_TRAIN * ROW_BYTES;                  //  1.0 MB
  unsigned* keyBase = (unsigned*)(testB + (size_t)NUM_TEST * ROW_BYTES); // 200 KB

  // Phase 1: pack + per-row sort-key bases.
  pack_rows_kernel<<<(NUM_TRAIN + 255) / 256, 256, 0, stream>>>(
      training_set, trainB, keyBase, NUM_TRAIN);
  pack_rows_kernel<<<(NUM_TEST + 255) / 256, 256, 0, stream>>>(
      test_set, testB, nullptr, NUM_TEST);

  // Phase 2: fused WMMA GEMM + top-3 + vote. One block per 16 test rows.
  knn_wmma_kernel<<<NUM_TEST / 16, 256, 0, stream>>>(testB, trainB, keyBase, out);
}
